// QKTranslution_78804059947347
// MI455X (gfx1250) — matrix operand order repaired and checked by reading, not compiled, not run on hardware
//
#include <hip/hip_runtime.h>
#include <hip/hip_bf16.h>

#define B_   2
#define L_   192
#define DIM_ 384
#define H_   8
#define DH_  48
#define INNER_ 384

typedef __attribute__((ext_vector_type(16))) _Float16 v16h;
typedef __attribute__((ext_vector_type(8)))  _Float16 v8h;
typedef __attribute__((ext_vector_type(8)))  float    v8f;

union AFrag16 { v16h v; v8h h[2]; };

// ---------------------------------------------------------------------------
// WMMA fragment helpers (V_WMMA_F32_16X16X32_F16, wave32)
// A (16x32 f16, row-major, leading dim ld): lane: m=lane&15, kb=(lane>>4)*8;
//   halves 0..7 -> K=kb..kb+7, halves 8..15 -> K=kb+16..kb+23
// ---------------------------------------------------------------------------
__device__ __forceinline__ v16h load_frag_a(const _Float16* tile, int ld, int lane) {
    const _Float16* p = tile + (size_t)(lane & 15) * ld + ((lane >> 4) << 3);
    AFrag16 f;
    f.h[0] = *(const v8h*)(p);
    f.h[1] = *(const v8h*)(p + 16);
    return f.v;
}

// B (32x16 f16) from COLUMN-MAJOR wT[n][k], leading dim ldk:
//   lane: n = n0+(lane&15), kb = k0 + (lane>>4)*8; halves 0..7 -> kb..kb+7, 8..15 -> kb+16..kb+23
__device__ __forceinline__ v16h load_frag_b(const _Float16* baseT, int ldk,
                                            int n0, int k0, int lane) {
    const _Float16* p = baseT + (size_t)(n0 + (lane & 15)) * ldk + k0 + ((lane >> 4) << 3);
    AFrag16 f;
    f.h[0] = *(const v8h*)(p);
    f.h[1] = *(const v8h*)(p + 16);
    return f.v;
}

__device__ __forceinline__ v8f wmma_f32_f16(v16h a, v16h b, v8f c) {
    return __builtin_amdgcn_wmma_f32_16x16x32_f16(false, a, false, b,
                                                  (short)0, c, false, false);
}

// C/D: vgpr r of lane holds m = r + 8*(lane>>4), n = lane&15
__device__ __forceinline__ void store_acc_f16(_Float16* dst, int ld, int lane, v8f acc) {
    int n  = lane & 15;
    int mh = (lane >> 4) << 3;
#pragma unroll
    for (int r = 0; r < 8; ++r)
        dst[(size_t)(mh + r) * ld + n] = (_Float16)acc[r];
}

// ---------------------------------------------------------------------------
// 1) LayerNorm (biased var) -> f16, one block per row
// ---------------------------------------------------------------------------
__global__ void qkt_ln_kernel(const float* __restrict__ x,
                              const float* __restrict__ gamma,
                              const float* __restrict__ beta,
                              _Float16* __restrict__ xn_h) {
    int row = blockIdx.x;                 // 0 .. B*L-1
    const float* xr = x + (size_t)row * DIM_;
    __shared__ float red[256];
    int tid = threadIdx.x;                // 128 threads
    float s = 0.f, s2 = 0.f;
    for (int d = tid; d < DIM_; d += 128) { float v = xr[d]; s += v; s2 += v * v; }
    red[tid] = s; red[128 + tid] = s2;
    __syncthreads();
    for (int off = 64; off > 0; off >>= 1) {
        if (tid < off) { red[tid] += red[tid + off]; red[128 + tid] += red[128 + tid + off]; }
        __syncthreads();
    }
    float mean = red[0] * (1.0f / DIM_);
    float var  = red[128] * (1.0f / DIM_) - mean * mean;
    float inv  = rsqrtf(var + 1e-5f);
    for (int d = tid; d < DIM_; d += 128) {
        float v = (xr[d] - mean) * inv * gamma[d] + beta[d];
        xn_h[(size_t)row * DIM_ + d] = (_Float16)v;
    }
}

// ---------------------------------------------------------------------------
// 2) f32 [K][N] row-major -> f16 col-major [N][K]
// ---------------------------------------------------------------------------
__global__ void qkt_cvtT_kernel(const float* __restrict__ w,
                                _Float16* __restrict__ wT, int K, int N) {
    int idx = blockIdx.x * 256 + threadIdx.x;
    if (idx >= K * N) return;
    int k = idx / N, n = idx % N;
    wT[(size_t)n * K + k] = (_Float16)w[idx];
}

// ---------------------------------------------------------------------------
// 3) V projection: v = xn @ w_v, stored transposed v_t[b][h][d][j]
//    wave task = (mt 0..23, head 0..7); 3 col tiles per task (one head)
// ---------------------------------------------------------------------------
__global__ __launch_bounds__(256)
void qkt_vproj_kernel(const _Float16* __restrict__ xn_h,
                      const _Float16* __restrict__ wvT,
                      _Float16* __restrict__ v_t) {
    int task = blockIdx.x * 8 + (threadIdx.x >> 5);
    int lane = threadIdx.x & 31;
    if (task >= 24 * H_) return;
    int mt = task >> 3;                   // row tile (16 rows of b*L)
    int h  = task & 7;                    // head
    const _Float16* A0 = xn_h + (size_t)mt * 16 * DIM_;
    const _Float16* W  = wvT + (size_t)h * DH_ * DIM_;   // cols h*48..h*48+47
    v8f a0 = {}, a1 = {}, a2 = {};
#pragma unroll 2
    for (int kk = 0; kk < DIM_; kk += 32) {
        v16h a  = load_frag_a(A0 + kk, DIM_, lane);
        v16h b0 = load_frag_b(W, DIM_, 0,  kk, lane);
        v16h b1 = load_frag_b(W, DIM_, 16, kk, lane);
        v16h b2 = load_frag_b(W, DIM_, 32, kk, lane);
        a0 = wmma_f32_f16(a, b0, a0);
        a1 = wmma_f32_f16(a, b1, a1);
        a2 = wmma_f32_f16(a, b2, a2);
    }
    int b  = mt / 12;                     // compile-time-friendly (mt<24)
    int j0 = (mt - b * 12) * 16;
    int mh = (lane >> 4) << 3;
    int nn = lane & 15;
    _Float16* vb = v_t + ((size_t)(b * H_ + h) * DH_) * L_;
#pragma unroll
    for (int r = 0; r < 8; ++r) {
        int j = j0 + mh + r;
        vb[(size_t)(0  + nn) * L_ + j] = (_Float16)a0[r];
        vb[(size_t)(16 + nn) * L_ + j] = (_Float16)a1[r];
        vb[(size_t)(32 + nn) * L_ + j] = (_Float16)a2[r];
    }
}

// ---------------------------------------------------------------------------
// 4) Main: per (offset t, head h): Q = xn @ u_q[L-1-t], K = xn @ u_k[t]
//    (needed rows only), then diagonal dot -> dots[b,h,i,i-t]
//    Each wave: one row tile, 3 col-tile accumulators (A fragment reused 3x)
// ---------------------------------------------------------------------------
__global__ __launch_bounds__(256)
void qkt_dots_kernel(const _Float16* __restrict__ xn_h,
                     const float* __restrict__ u_q,
                     const float* __restrict__ u_k,
                     float* __restrict__ dots) {
    const int t = blockIdx.x;             // offset 0..L-1
    const int h = blockIdx.y;             // head
    extern __shared__ _Float16 lds[];
    _Float16* wqT = lds;                  // [48][384] col-major slice of u_q[L-1-t]
    _Float16* wkT = wqT + DH_ * DIM_;     // [48][384]
    _Float16* Qr  = wkT + DH_ * DIM_;     // [192][48] f16 results
    _Float16* Kr  = Qr + L_ * DH_;        // [192][48]

    int tid = threadIdx.x, lane = tid & 31, wave = tid >> 5;

    // Stage weight slices for this head: wT[n][k] = u[r][k][48h+n]
    const float* uq = u_q + ((size_t)(L_ - 1 - t) * DIM_) * INNER_ + h * DH_;
    const float* uk = u_k + ((size_t)t * DIM_) * INNER_ + h * DH_;
    for (int idx = tid; idx < DH_ * DIM_; idx += 256) {
        int k = idx / DH_, n = idx % DH_;
        wqT[(size_t)n * DIM_ + k] = (_Float16)uq[(size_t)k * INNER_ + n];
        wkT[(size_t)n * DIM_ + k] = (_Float16)uk[(size_t)k * INNER_ + n];
    }
    __syncthreads();

    const int qt0 = t >> 4;               // first Q row tile (rows >= t)
    const int nq  = (L_ / 16) - qt0;      // # Q row tiles
    const int nk  = (L_ - t + 15) >> 4;   // # K row tiles (rows < L-t)
    const float scale = rsqrtf((float)DH_);

    for (int b = 0; b < B_; ++b) {
        const _Float16* X = xn_h + (size_t)b * L_ * DIM_;
        for (int rt = wave; rt < nq + nk; rt += 8) {
            bool isQ = rt < nq;
            int rowt = isQ ? (qt0 + rt) : (rt - nq);
            const _Float16* W = isQ ? wqT : wkT;
            _Float16*       R = isQ ? Qr  : Kr;
            const _Float16* A0 = X + (size_t)rowt * 16 * DIM_;
            v8f c0 = {}, c1 = {}, c2 = {};
#pragma unroll 2
            for (int kk = 0; kk < DIM_; kk += 32) {
                v16h a  = load_frag_a(A0 + kk, DIM_, lane);
                v16h b0 = load_frag_b(W, DIM_, 0,  kk, lane);
                v16h b1 = load_frag_b(W, DIM_, 16, kk, lane);
                v16h b2 = load_frag_b(W, DIM_, 32, kk, lane);
                c0 = wmma_f32_f16(a, b0, c0);
                c1 = wmma_f32_f16(a, b1, c1);
                c2 = wmma_f32_f16(a, b2, c2);
            }
            _Float16* Rt = R + (size_t)rowt * 16 * DH_;
            store_acc_f16(Rt + 0,  DH_, lane, c0);
            store_acc_f16(Rt + 16, DH_, lane, c1);
            store_acc_f16(Rt + 32, DH_, lane, c2);
        }
        __syncthreads();

        // Diagonal dot: dots[b,h,i,i-t] = scale * <Qr[i,:], Kr[i-t,:]>
        for (int i = t + tid; i < L_; i += 256) {
            const _Float16* q  = Qr + (size_t)i * DH_;
            const _Float16* kv = Kr + (size_t)(i - t) * DH_;
            float s = 0.f;
#pragma unroll
            for (int d = 0; d < DH_; ++d) s += (float)q[d] * (float)kv[d];
            dots[(((size_t)b * H_ + h) * L_ + i) * L_ + (i - t)] = s * scale;
        }
        __syncthreads();
    }
}

// ---------------------------------------------------------------------------
// 5) Causal softmax over j<=i, f16 output, zero above diagonal
// ---------------------------------------------------------------------------
__global__ void qkt_softmax_kernel(const float* __restrict__ dots,
                                   _Float16* __restrict__ attn_h) {
    int bhi = blockIdx.x;                 // ((b*H+h)*L + i)
    int i   = bhi % L_;
    const float* row = dots + (size_t)bhi * L_;
    _Float16* arow = attn_h + (size_t)bhi * L_;
    __shared__ float red[128];
    int tid = threadIdx.x;                // 128 threads
    float m = -3.4e38f;
    for (int j = tid; j <= i; j += 128) m = fmaxf(m, row[j]);
    red[tid] = m; __syncthreads();
    for (int off = 64; off > 0; off >>= 1) {
        if (tid < off) red[tid] = fmaxf(red[tid], red[tid + off]);
        __syncthreads();
    }
    m = red[0]; __syncthreads();
    float s = 0.f;
    for (int j = tid; j <= i; j += 128) s += __expf(row[j] - m);
    red[tid] = s; __syncthreads();
    for (int off = 64; off > 0; off >>= 1) {
        if (tid < off) red[tid] += red[tid + off];
        __syncthreads();
    }
    float inv = 1.0f / red[0];
    for (int j = tid; j < L_; j += 128)
        arow[j] = (_Float16)((j <= i) ? __expf(row[j] - m) * inv : 0.0f);
}

// ---------------------------------------------------------------------------
// 6) ctx = attn @ v (per b,h), K=192; wave task = (bh, mt), 3 col tiles
// ---------------------------------------------------------------------------
__global__ __launch_bounds__(256)
void qkt_attnv_kernel(const _Float16* __restrict__ attn_h,
                      const _Float16* __restrict__ v_t,
                      _Float16* __restrict__ ctx_h) {
    int task = blockIdx.x * 8 + (threadIdx.x >> 5);
    int lane = threadIdx.x & 31;
    if (task >= B_ * H_ * 12) return;
    int mt = task % 12;
    int bh = task / 12;
    const _Float16* A0 = attn_h + ((size_t)bh * L_ + mt * 16) * L_;
    const _Float16* B0 = v_t + (size_t)bh * DH_ * L_;   // col-major [d][j], ldk=L
    v8f c0 = {}, c1 = {}, c2 = {};
#pragma unroll 2
    for (int kk = 0; kk < L_; kk += 32) {
        v16h a  = load_frag_a(A0 + kk, L_, lane);
        v16h b0 = load_frag_b(B0, L_, 0,  kk, lane);
        v16h b1 = load_frag_b(B0, L_, 16, kk, lane);
        v16h b2 = load_frag_b(B0, L_, 32, kk, lane);
        c0 = wmma_f32_f16(a, b0, c0);
        c1 = wmma_f32_f16(a, b1, c1);
        c2 = wmma_f32_f16(a, b2, c2);
    }
    int b = bh >> 3, h = bh & 7;
    int nn = lane & 15;
    int ibase = mt * 16 + ((lane >> 4) << 3);
    _Float16* cb = ctx_h + ((size_t)b * L_) * INNER_ + h * DH_;
#pragma unroll
    for (int r = 0; r < 8; ++r) {
        size_t ro = (size_t)(ibase + r) * INNER_;
        cb[ro + 0  + nn] = (_Float16)c0[r];
        cb[ro + 16 + nn] = (_Float16)c1[r];
        cb[ro + 32 + nn] = (_Float16)c2[r];
    }
}

// ---------------------------------------------------------------------------
// 7) out = ctx @ w_o + b_o (fp32); wave task = (mt 0..23, ng 0..5), 4 col tiles
// ---------------------------------------------------------------------------
__global__ __launch_bounds__(256)
void qkt_outproj_kernel(const _Float16* __restrict__ ctx_h,
                        const _Float16* __restrict__ woT,
                        const float* __restrict__ b_o,
                        float* __restrict__ out) {
    int task = blockIdx.x * 8 + (threadIdx.x >> 5);
    int lane = threadIdx.x & 31;
    if (task >= 24 * 6) return;
    int mt = task / 6;
    int n0 = (task % 6) * 64;             // 4 col tiles of 16
    const _Float16* A0 = ctx_h + (size_t)mt * 16 * INNER_;
    v8f c0 = {}, c1 = {}, c2 = {}, c3 = {};
#pragma unroll 2
    for (int kk = 0; kk < INNER_; kk += 32) {
        v16h a  = load_frag_a(A0 + kk, INNER_, lane);
        v16h b0 = load_frag_b(woT, INNER_, n0 + 0,  kk, lane);
        v16h b1 = load_frag_b(woT, INNER_, n0 + 16, kk, lane);
        v16h b2 = load_frag_b(woT, INNER_, n0 + 32, kk, lane);
        v16h b3 = load_frag_b(woT, INNER_, n0 + 48, kk, lane);
        c0 = wmma_f32_f16(a, b0, c0);
        c1 = wmma_f32_f16(a, b1, c1);
        c2 = wmma_f32_f16(a, b2, c2);
        c3 = wmma_f32_f16(a, b3, c3);
    }
    int nn = lane & 15;
    int mbase = mt * 16 + ((lane >> 4) << 3);
    float bias0 = b_o[n0 + 0  + nn];
    float bias1 = b_o[n0 + 16 + nn];
    float bias2 = b_o[n0 + 32 + nn];
    float bias3 = b_o[n0 + 48 + nn];
#pragma unroll
    for (int r = 0; r < 8; ++r) {
        size_t ro = (size_t)(mbase + r) * DIM_ + n0;
        out[ro + 0  + nn] = c0[r] + bias0;
        out[ro + 16 + nn] = c1[r] + bias1;
        out[ro + 32 + nn] = c2[r] + bias2;
        out[ro + 48 + nn] = c3[r] + bias3;
    }
}

// ---------------------------------------------------------------------------
extern "C" void kernel_launch(void* const* d_in, const int* in_sizes, int n_in,
                              void* d_out, int out_size, void* d_ws, size_t ws_size,
                              hipStream_t stream) {
    const float* x     = (const float*)d_in[0];
    const float* gamma = (const float*)d_in[1];
    const float* beta  = (const float*)d_in[2];
    const float* u_q   = (const float*)d_in[3];
    const float* u_k   = (const float*)d_in[4];
    const float* w_v   = (const float*)d_in[5];
    const float* w_o   = (const float*)d_in[6];
    const float* b_o   = (const float*)d_in[7];
    float* out = (float*)d_out;

    // workspace layout (bytes)
    char* ws = (char*)d_ws;
    _Float16* xn_h   = (_Float16*)(ws + 0);              // 294912 B
    _Float16* wvT    = (_Float16*)(ws + 294912);         // 294912 B
    _Float16* woT    = (_Float16*)(ws + 589824);         // 294912 B
    _Float16* v_t    = (_Float16*)(ws + 884736);         // 294912 B
    _Float16* attn_h = (_Float16*)(ws + 1179648);        // 1179648 B
    _Float16* ctx_h  = (_Float16*)(ws + 2359296);        // 294912 B
    float*    dots   = (float*)   (ws + 2654208);        // 2359296 B (total ~4.8 MB)

    // 1) LayerNorm -> f16
    qkt_ln_kernel<<<B_ * L_, 128, 0, stream>>>(x, gamma, beta, xn_h);

    // 2) weight transposes to f16 col-major
    int nW = DIM_ * INNER_;
    qkt_cvtT_kernel<<<(nW + 255) / 256, 256, 0, stream>>>(w_v, wvT, DIM_, INNER_);
    qkt_cvtT_kernel<<<(nW + 255) / 256, 256, 0, stream>>>(w_o, woT, INNER_, DIM_);

    // 3) V projection
    qkt_vproj_kernel<<<(24 * H_ + 7) / 8, 256, 0, stream>>>(xn_h, wvT, v_t);

    // 4) per-offset Q/K GEMMs + diagonal dots (dominant kernel)
    const int dotsLds = (2 * DH_ * DIM_ + 2 * L_ * DH_) * (int)sizeof(_Float16); // 110592 B
    hipFuncSetAttribute((const void*)qkt_dots_kernel,
                        hipFuncAttributeMaxDynamicSharedMemorySize, dotsLds);
    qkt_dots_kernel<<<dim3(L_, H_), 256, dotsLds, stream>>>(xn_h, u_q, u_k, dots);

    // 5) causal softmax
    qkt_softmax_kernel<<<B_ * H_ * L_, 128, 0, stream>>>(dots, attn_h);

    // 6) attn @ v
    qkt_attnv_kernel<<<(B_ * H_ * 12 + 7) / 8, 256, 0, stream>>>(attn_h, v_t, ctx_h);

    // 7) output projection + bias
    qkt_outproj_kernel<<<(24 * 6 + 7) / 8, 256, 0, stream>>>(ctx_h, woT, b_o, out);
}
